// VisionTransformer_63952063037625
// MI455X (gfx1250) — matrix operand order repaired and checked by reading, not compiled, not run on hardware
//
#include <hip/hip_runtime.h>
#include <hip/hip_bf16.h>
#include <math.h>

#define B_      32
#define NTOK_   197
#define CDIM    768
#define HEADS_  12
#define HD_     64
#define DEPTH_  12
#define NPATCH  196
#define MTOK    (B_*NTOK_)    // 6304 = 197*32
#define MPATCH  (B_*NPATCH)   // 6272 = 196*32

typedef __attribute__((ext_vector_type(16))) __bf16 v16bf;
typedef __attribute__((ext_vector_type(8)))  __bf16 bf16x8;
typedef __attribute__((ext_vector_type(8)))  float  v8f;
typedef __attribute__((ext_vector_type(4)))  float  f32x4;

// ---------------------------------------------------------------------------
// Pack f32 weights [N,K] into bf16 WMMA B-fragment order:
//   out[((nt*(K/32)+ks)*32 + lane)*16 + e] = W[(nt*16 + lane&15)*K + ks*32 + (lane>>4)*16 + e]
// One wave per 16x32 fragment block (1 KB out, contiguous).
// ---------------------------------------------------------------------------
__global__ __launch_bounds__(256) void pack_weight_bf16(
    const float* __restrict__ W, __bf16* __restrict__ out, int N, int K)
{
  const int lane = threadIdx.x & 31;
  const int wave = threadIdx.x >> 5;
  const int ksteps = K >> 5;
  const int total = (N >> 4) * ksteps;
  const int fid = blockIdx.x * 8 + wave;
  if (fid >= total) return;
  const int nt = fid / ksteps, ks = fid % ksteps;
  const float* src = W + (size_t)(nt * 16 + (lane & 15)) * K + ks * 32 + (lane >> 4) * 16;
  f32x4 q0 = ((const f32x4*)src)[0];
  f32x4 q1 = ((const f32x4*)src)[1];
  f32x4 q2 = ((const f32x4*)src)[2];
  f32x4 q3 = ((const f32x4*)src)[3];
  bf16x8 b0, b1;
#pragma unroll
  for (int e = 0; e < 4; ++e) {
    b0[e]     = (__bf16)q0[e];
    b0[e + 4] = (__bf16)q1[e];
    b1[e]     = (__bf16)q2[e];
    b1[e + 4] = (__bf16)q3[e];
  }
  __bf16* dst = out + ((size_t)fid * 32 + lane) * 16;
  *(bf16x8*)dst       = b0;
  *(bf16x8*)(dst + 8) = b1;
}

// ---------------------------------------------------------------------------
// Tiled bf16 WMMA GEMM: C[M,N] = A[M,K] (bf16 row-major) * Wp (packed bf16)
// Each wave computes a 32x64 tile: 8 x v_wmma_f32_16x16x32_bf16 per K-step.
// epi: 0 = bias -> f32 ; 1 = bias + residual-add into f32 ; 2 = bias+GELU -> bf16
// ---------------------------------------------------------------------------
__global__ __launch_bounds__(256) void gemm_bf16_wmma(
    const __bf16* __restrict__ A, const __bf16* __restrict__ Wp,
    const float* __restrict__ bias, int M, int N, int K,
    int epi, float* __restrict__ outF, __bf16* __restrict__ outB,
    float* __restrict__ resid)
{
  const int lane = threadIdx.x & 31;
  const int wave = threadIdx.x >> 5;
  const int nstrips = N >> 6;
  const int total = (M >> 5) * nstrips;
  const int gid = blockIdx.x * 8 + wave;
  if (gid >= total) return;                 // whole-wave uniform exit (EXEC all-1 for WMMA)
  const int mtile  = gid / nstrips;         // 32-row tile
  const int nbase  = (gid % nstrips) << 6;
  const int ntile0 = nbase >> 4;
  const int ksteps = K >> 5;
  const int hi  = lane >> 4;
  const int l15 = lane & 15;

  v8f acc[8] = {};

  // A fragment (contiguous k = 16*hi + e, matching packed B): per lane two 16B chunks.
  const __bf16* arow0 = A + (size_t)((mtile << 5) + l15) * K + hi * 16;
  const __bf16* arow1 = arow0 + (size_t)16 * K;
  // Packed B: contiguous 1 KB per (ntile, kstep) fragment block.
  const __bf16* bbase = Wp + (size_t)ntile0 * ksteps * 512 + lane * 16;

  for (int ks = 0; ks < ksteps; ++ks) {
    const int k0 = ks << 5;
    if (k0 + 32 < K) {
      __builtin_prefetch(arow0 + k0 + 32, 0, 1);   // global_prefetch_b8
      __builtin_prefetch(arow1 + k0 + 32, 0, 1);
    }
    bf16x8 a00 = *(const bf16x8*)(arow0 + k0);
    bf16x8 a01 = *(const bf16x8*)(arow0 + k0 + 8);
    bf16x8 a10 = *(const bf16x8*)(arow1 + k0);
    bf16x8 a11 = *(const bf16x8*)(arow1 + k0 + 8);
    v16bf af0, af1;
#pragma unroll
    for (int e = 0; e < 8; ++e) {
      af0[e] = a00[e]; af0[e + 8] = a01[e];
      af1[e] = a10[e]; af1[e + 8] = a11[e];
    }
#pragma unroll
    for (int t = 0; t < 4; ++t) {
      const __bf16* bp = bbase + ((size_t)t * ksteps + ks) * 512;
      bf16x8 b0 = *(const bf16x8*)bp;
      bf16x8 b1 = *(const bf16x8*)(bp + 8);
      v16bf bf;
#pragma unroll
      for (int e = 0; e < 8; ++e) { bf[e] = b0[e]; bf[e + 8] = b1[e]; }
      acc[t]     = __builtin_amdgcn_wmma_f32_16x16x32_bf16(
          false, af0, false, bf, (short)0, acc[t],     false, false);
      acc[t + 4] = __builtin_amdgcn_wmma_f32_16x16x32_bf16(
          false, af1, false, bf, (short)0, acc[t + 4], false, false);
    }
  }

  // Epilogue. C/D layout: lane half hi -> rows M = r + 8*hi, N = l15.
#pragma unroll
  for (int t = 0; t < 4; ++t) {
    const int n = nbase + t * 16 + l15;
    const float bv = bias ? bias[n] : 0.f;
#pragma unroll
    for (int half = 0; half < 2; ++half) {
      const int mb = (mtile << 5) + half * 16 + hi * 8;
#pragma unroll
      for (int r = 0; r < 8; ++r) {
        const int m = mb + r;
        float v = acc[t + 4 * half][r] + bv;
        if (epi == 0) {
          outF[(size_t)m * N + n] = v;
        } else if (epi == 1) {
          resid[(size_t)m * N + n] += v;
        } else {
          float g = 0.5f * v * (1.f + erff(v * 0.70710678118654752f));
          outB[(size_t)m * N + n] = (__bf16)g;
        }
      }
    }
  }
}

// ---------------------------------------------------------------------------
// im2col: x[B,3,224,224] -> bf16 [6272, 768], K index = c*256 + i*16 + j
// ---------------------------------------------------------------------------
__global__ __launch_bounds__(256) void im2col_kernel(const float* __restrict__ x,
                                                     __bf16* __restrict__ out)
{
  const int m = blockIdx.x;
  const int b = m / NPATCH;
  const int t = m % NPATCH;
  const int py = t / 14, px = t % 14;
  for (int kk = threadIdx.x; kk < CDIM; kk += 256) {
    const int c = kk >> 8, rem = kk & 255, i = rem >> 4, j = rem & 15;
    const float v = x[(((size_t)b * 3 + c) * 224 + (py * 16 + i)) * 224 + (px * 16 + j)];
    out[(size_t)m * CDIM + kk] = (__bf16)v;
  }
}

// tokens = concat(cls, patches) + pos  (patch bias already fused in GEMM)
__global__ __launch_bounds__(256) void assemble_tokens(const float* __restrict__ p,
    const float* __restrict__ cls, const float* __restrict__ pos,
    float* __restrict__ tokens)
{
  const int row = blockIdx.x;
  const int b = row / NTOK_, n = row % NTOK_;
  for (int c = threadIdx.x; c < CDIM; c += 256) {
    float v = (n == 0) ? cls[c] : p[((size_t)b * NPATCH + (n - 1)) * CDIM + c];
    tokens[(size_t)row * CDIM + c] = v + pos[(size_t)n * CDIM + c];
  }
}

// ---------------------------------------------------------------------------
// LayerNorm over 768: one 256-thread block per row, 3 elems/thread.
// ---------------------------------------------------------------------------
__global__ __launch_bounds__(256) void layernorm_kernel(const float* __restrict__ in,
    const float* __restrict__ w, const float* __restrict__ b,
    __bf16* __restrict__ outB, float* __restrict__ outF)
{
  __shared__ float red[256];
  const int row = blockIdx.x, t = threadIdx.x;
  const float* rp = in + (size_t)row * CDIM;
  float v0 = rp[t], v1 = rp[t + 256], v2 = rp[t + 512];
  red[t] = v0 + v1 + v2;
  __syncthreads();
  for (int o = 128; o > 0; o >>= 1) { if (t < o) red[t] += red[t + o]; __syncthreads(); }
  const float mu = red[0] * (1.f / 768.f);
  __syncthreads();
  float d0 = v0 - mu, d1 = v1 - mu, d2 = v2 - mu;
  red[t] = d0 * d0 + d1 * d1 + d2 * d2;
  __syncthreads();
  for (int o = 128; o > 0; o >>= 1) { if (t < o) red[t] += red[t + o]; __syncthreads(); }
  const float rstd = rsqrtf(red[0] * (1.f / 768.f) + 1e-6f);
  float o0 = d0 * rstd * w[t]       + b[t];
  float o1 = d1 * rstd * w[t + 256] + b[t + 256];
  float o2 = d2 * rstd * w[t + 512] + b[t + 512];
  if (outB) {
    __bf16* op = outB + (size_t)row * CDIM;
    op[t] = (__bf16)o0; op[t + 256] = (__bf16)o1; op[t + 512] = (__bf16)o2;
  } else {
    float* op = outF + (size_t)row * CDIM;
    op[t] = o0; op[t + 256] = o1; op[t + 512] = o2;
  }
}

__global__ void zero_f32(float* __restrict__ p, int n)
{
  int i = blockIdx.x * 256 + threadIdx.x;
  if (i < n) p[i] = 0.f;
}

// ---------------------------------------------------------------------------
// Probe attention: softmax(q_cls . k * scale) per (b,h); accumulate head-mean
// of patch probs into cls_scores[B,196]. One block per (b,h).
// ---------------------------------------------------------------------------
__global__ __launch_bounds__(256) void probe_kernel(const float* __restrict__ qkv,
                                                    float* __restrict__ cls_scores)
{
  __shared__ float q[HD_];
  __shared__ float red[256];
  const int bh = blockIdx.x, b = bh / HEADS_, h = bh % HEADS_;
  const int t = threadIdx.x;
  if (t < HD_) q[t] = qkv[(size_t)(b * NTOK_) * 3 * CDIM + h * HD_ + t];
  __syncthreads();
  float s = -1e30f;
  if (t < NTOK_) {
    const float* krow = qkv + (size_t)(b * NTOK_ + t) * 3 * CDIM + CDIM + h * HD_;
    float acc = 0.f;
    for (int d = 0; d < HD_; ++d) acc += q[d] * krow[d];
    s = acc * 0.125f;
  }
  red[t] = s; __syncthreads();
  for (int o = 128; o > 0; o >>= 1) { if (t < o) red[t] = fmaxf(red[t], red[t + o]); __syncthreads(); }
  const float mx = red[0]; __syncthreads();
  float p = (t < NTOK_) ? __expf(s - mx) : 0.f;
  red[t] = p; __syncthreads();
  for (int o = 128; o > 0; o >>= 1) { if (t < o) red[t] += red[t + o]; __syncthreads(); }
  const float inv = 1.f / red[0];
  if (t >= 1 && t < NTOK_)
    atomicAdd(&cls_scores[b * NPATCH + (t - 1)], p * inv * (1.f / HEADS_));
}

// Top-5 of cls_scores -> sel mask [B,197]; re-zeroes scores for next layer.
__global__ __launch_bounds__(256) void topk_sel_kernel(float* __restrict__ cls_scores,
                                                       float* __restrict__ sel)
{
  __shared__ float s[NPATCH];
  __shared__ float selm[NTOK_];
  __shared__ float rv[256];
  __shared__ int   ri[256];
  const int b = blockIdx.x, t = threadIdx.x;
  if (t < NPATCH) s[t] = cls_scores[b * NPATCH + t];
  if (t < NTOK_)  selm[t] = 0.f;
  __syncthreads();
  for (int round = 0; round < 5; ++round) {
    rv[t] = (t < NPATCH) ? s[t] : -1e30f;
    ri[t] = t;
    __syncthreads();
    for (int o = 128; o > 0; o >>= 1) {
      if (t < o) {
        if (rv[t + o] > rv[t] || (rv[t + o] == rv[t] && ri[t + o] < ri[t])) {
          rv[t] = rv[t + o]; ri[t] = ri[t + o];
        }
      }
      __syncthreads();
    }
    if (t == 0) { int w = ri[0]; s[w] = -1e30f; selm[w + 1] = 1.f; }
    __syncthreads();
  }
  if (t < NTOK_)  sel[b * NTOK_ + t] = selm[t];
  if (t < NPATCH) cls_scores[b * NPATCH + t] = 0.f;
}

// ---------------------------------------------------------------------------
// Full attention with prompt-modified K/V; one block per (b,h), rows serial.
// Output written as bf16 [6304, 768] (direct A-input of the proj GEMM).
// ---------------------------------------------------------------------------
__global__ __launch_bounds__(256) void attn_kernel(const float* __restrict__ qkv,
    const float* __restrict__ sel,
    const float* __restrict__ prompt_k, const float* __restrict__ prompt_v,
    const float* __restrict__ pk_param, const float* __restrict__ pv_param,
    __bf16* __restrict__ attn_out)
{
  __shared__ float pk[HD_], pv[HD_], prk[HD_], prv[HD_];
  __shared__ float q[HD_];
  __shared__ float probs[NTOK_];
  __shared__ float selsh[NTOK_];
  __shared__ float red[256];
  const int bh = blockIdx.x, b = bh / HEADS_, h = bh % HEADS_;
  const int t = threadIdx.x;
  if (t < HD_) {
    pk[t]  = pk_param[h * HD_ + t];
    pv[t]  = pv_param[h * HD_ + t];
    prk[t] = prompt_k[h * HD_ + t];
    prv[t] = prompt_v[h * HD_ + t];
  }
  if (t < NTOK_) selsh[t] = sel[b * NTOK_ + t];
  __syncthreads();

  for (int qi = 0; qi < NTOK_; ++qi) {
    if (t < HD_) q[t] = qkv[(size_t)(b * NTOK_ + qi) * 3 * CDIM + h * HD_ + t];
    __syncthreads();

    float s = -1e30f;
    if (t < NTOK_) {
      const float* krow = qkv + (size_t)(b * NTOK_ + t) * 3 * CDIM + CDIM + h * HD_;
      const float kf = selsh[t];
      float acc = 0.f;
      if (t == 0) { for (int d = 0; d < HD_; ++d) acc += q[d] * (krow[d] + prk[d] + kf * pk[d]); }
      else        { for (int d = 0; d < HD_; ++d) acc += q[d] * (krow[d] + kf * pk[d]); }
      s = acc * 0.125f;
    }
    red[t] = s; __syncthreads();
    for (int o = 128; o > 0; o >>= 1) { if (t < o) red[t] = fmaxf(red[t], red[t + o]); __syncthreads(); }
    const float mx = red[0]; __syncthreads();
    float p = (t < NTOK_) ? __expf(s - mx) : 0.f;
    if (t < NTOK_) probs[t] = p;
    red[t] = p; __syncthreads();
    for (int o = 128; o > 0; o >>= 1) { if (t < o) red[t] += red[t + o]; __syncthreads(); }
    const float inv = 1.f / red[0];
    __syncthreads();

    const int d = t & 63, chunk = t >> 6;
    float partial = 0.f;
    for (int j = chunk; j < NTOK_; j += 4) {
      const float* vrow = qkv + (size_t)(b * NTOK_ + j) * 3 * CDIM + 2 * CDIM + h * HD_;
      float vm = vrow[d] + ((j == 0) ? prv[d] : 0.f) + selsh[j] * pv[d];
      partial += probs[j] * vm;
    }
    red[t] = partial; __syncthreads();
    if (t < HD_) {
      float o = (red[t] + red[t + 64] + red[t + 128] + red[t + 192]) * inv;
      attn_out[(size_t)(b * NTOK_ + qi) * CDIM + h * HD_ + t] = (__bf16)o;
    }
    __syncthreads();
  }
}

// ---------------------------------------------------------------------------
extern "C" void kernel_launch(void* const* d_in, const int* in_sizes, int n_in,
                              void* d_out, int out_size, void* d_ws, size_t ws_size,
                              hipStream_t stream)
{
  (void)in_sizes; (void)n_in; (void)out_size; (void)ws_size;
  const float* x        = (const float*)d_in[0];
  const float* patch_w  = (const float*)d_in[1];
  const float* patch_b  = (const float*)d_in[2];
  const float* cls_tok  = (const float*)d_in[3];
  const float* pos      = (const float*)d_in[4];
  const float* ln1_w    = (const float*)d_in[5];
  const float* ln1_b    = (const float*)d_in[6];
  const float* qkv_w    = (const float*)d_in[7];
  const float* qkv_b    = (const float*)d_in[8];
  const float* proj_w   = (const float*)d_in[9];
  const float* proj_b   = (const float*)d_in[10];
  const float* pk_param = (const float*)d_in[11];
  const float* pv_param = (const float*)d_in[12];
  const float* prompt_k = (const float*)d_in[13];
  const float* prompt_v = (const float*)d_in[14];
  const float* ln2_w    = (const float*)d_in[15];
  const float* ln2_b    = (const float*)d_in[16];
  const float* fc1_w    = (const float*)d_in[17];
  const float* fc1_b    = (const float*)d_in[18];
  const float* fc2_w    = (const float*)d_in[19];
  const float* fc2_b    = (const float*)d_in[20];
  const float* norm_w   = (const float*)d_in[21];
  const float* norm_b   = (const float*)d_in[22];
  // d_in[23] = k_patches == 5 (compile-time constant here)

  char* ws = (char*)d_ws;
  size_t off = 0;
  auto alloc = [&](size_t bytes) {
    void* p = ws + off;
    off += (bytes + 255) & ~(size_t)255;
    return p;
  };
  float*  tokens = (float*) alloc((size_t)MTOK * CDIM * 4);         // 19.4 MB
  __bf16* hbf    = (__bf16*)alloc((size_t)MTOK * CDIM * 2);         //  9.7 MB
  __bf16* attnbf = (__bf16*)alloc((size_t)MTOK * CDIM * 2);         //  9.7 MB
  float*  qkvf   = (float*) alloc((size_t)MTOK * 3 * CDIM * 4);     // 58.1 MB
  __bf16* mbf    = (__bf16*)alloc((size_t)MTOK * 4 * CDIM * 2);     // 38.7 MB
  __bf16* wbf    = (__bf16*)alloc((size_t)4 * CDIM * CDIM * 2);     //  4.7 MB packed weights
  float*  scores = (float*) alloc((size_t)B_ * NPATCH * 4);
  float*  sel    = (float*) alloc((size_t)B_ * NTOK_ * 4);

  auto gemm = [&](const __bf16* A, const float* Wsrc, const float* bias,
                  int M, int N, int K, int epi, float* outF, __bf16* outB, float* resid) {
    int frags  = (N / 16) * (K / 32);
    pack_weight_bf16<<<dim3((frags + 7) / 8), dim3(256), 0, stream>>>(Wsrc, wbf, N, K);
    int total  = (M / 32) * (N / 64);
    gemm_bf16_wmma<<<dim3((total + 7) / 8), dim3(256), 0, stream>>>(
        A, wbf, bias, M, N, K, epi, outF, outB, resid);
  };

  zero_f32<<<dim3((B_ * NPATCH + 255) / 256), dim3(256), 0, stream>>>(scores, B_ * NPATCH);

  // Patch embedding: im2col + WMMA GEMM (6272 x 768 x 768), bias fused.
  im2col_kernel<<<dim3(MPATCH), dim3(256), 0, stream>>>(x, hbf);
  gemm(hbf, patch_w, patch_b, MPATCH, CDIM, CDIM, 0, qkvf, nullptr, nullptr);
  assemble_tokens<<<dim3(MTOK), dim3(256), 0, stream>>>(qkvf, cls_tok, pos, tokens);

  for (int l = 0; l < DEPTH_; ++l) {
    layernorm_kernel<<<dim3(MTOK), dim3(256), 0, stream>>>(
        tokens, ln1_w + l * CDIM, ln1_b + l * CDIM, hbf, (float*)nullptr);
    gemm(hbf, qkv_w + (size_t)l * 3 * CDIM * CDIM, qkv_b + l * 3 * CDIM,
         MTOK, 3 * CDIM, CDIM, 0, qkvf, nullptr, nullptr);
    probe_kernel<<<dim3(B_ * HEADS_), dim3(256), 0, stream>>>(qkvf, scores);
    topk_sel_kernel<<<dim3(B_), dim3(256), 0, stream>>>(scores, sel);
    attn_kernel<<<dim3(B_ * HEADS_), dim3(256), 0, stream>>>(
        qkvf, sel, prompt_k + l * HEADS_ * HD_, prompt_v + l * HEADS_ * HD_,
        pk_param + l * HEADS_ * HD_, pv_param + l * HEADS_ * HD_, attnbf);
    gemm(attnbf, proj_w + (size_t)l * CDIM * CDIM, proj_b + l * CDIM,
         MTOK, CDIM, CDIM, 1, nullptr, nullptr, tokens);
    layernorm_kernel<<<dim3(MTOK), dim3(256), 0, stream>>>(
        tokens, ln2_w + l * CDIM, ln2_b + l * CDIM, hbf, (float*)nullptr);
    gemm(hbf, fc1_w + (size_t)l * 4 * CDIM * CDIM, fc1_b + l * 4 * CDIM,
         MTOK, 4 * CDIM, CDIM, 2, nullptr, mbf, nullptr);
    gemm(mbf, fc2_w + (size_t)l * CDIM * 4 * CDIM, fc2_b + l * CDIM,
         MTOK, CDIM, 4 * CDIM, 1, nullptr, nullptr, tokens);
  }

  layernorm_kernel<<<dim3(MTOK), dim3(256), 0, stream>>>(
      tokens, norm_w, norm_b, (__bf16*)nullptr, (float*)d_out);
}
